// EchoAttention_16080357556563
// MI455X (gfx1250) — matrix operand order repaired and checked by reading, not compiled, not run on hardware
//
#include <hip/hip_runtime.h>
#include <hip/hip_bf16.h>
#include <math.h>

// ---------------------------------------------------------------------------
// Flash attention for B=2, S=8192, D=128 (fp32 in/out, f16 WMMA compute).
// gfx1250 (CDNA5): wave32, v_wmma_f32_16x16x32_f16, DPP softmax reductions,
// packed-f32 accumulator rescale.
// ---------------------------------------------------------------------------

#define BATCH 2
#define SEQ   8192
#define DIM   128

#define KV_TILE 64          // keys per iteration (4 x 16-key subtiles)
#define WAVES_PER_BLOCK 8   // 8 waves * 16 q-rows = 128 q-rows per block
#define BLOCK_THREADS (WAVES_PER_BLOCK * 32)
#define Q_ROWS_PER_BLOCK (WAVES_PER_BLOCK * 16)

typedef _Float16 h2  __attribute__((ext_vector_type(2)));
typedef _Float16 h8  __attribute__((ext_vector_type(8)));
typedef _Float16 h16 __attribute__((ext_vector_type(16)));
typedef float    f8  __attribute__((ext_vector_type(8)));

union H16 {
    h16 v;
    h8  h[2];
    _Float16 e[16];
};

// LDS strides (halves): every b128 fragment load must be 16B aligned.
#define K_STRIDE  136   // 272 B per row
#define VT_STRIDE 72    // 144 B per row

// ---- DPP butterfly reduction within each 16-lane row (wave32) -------------
// quad_perm(1,0,3,2)=0xB1, quad_perm(2,3,0,1)=0x4E,
// ROW_HALF_MIRROR=0x141 (lane^7), ROW_MIRROR=0x140 (lane^15).
template <int CTRL>
__device__ __forceinline__ float dpp_bcast(float x) {
    int xi = __builtin_bit_cast(int, x);
    int yi = __builtin_amdgcn_update_dpp(xi, xi, CTRL, 0xf, 0xf, true);
    return __builtin_bit_cast(float, yi);
}
__device__ __forceinline__ float row16_max(float x) {
    x = fmaxf(x, dpp_bcast<0xB1>(x));
    x = fmaxf(x, dpp_bcast<0x4E>(x));
    x = fmaxf(x, dpp_bcast<0x141>(x));
    x = fmaxf(x, dpp_bcast<0x140>(x));
    return x;
}
__device__ __forceinline__ float row16_sum(float x) {
    x += dpp_bcast<0xB1>(x);
    x += dpp_bcast<0x4E>(x);
    x += dpp_bcast<0x141>(x);
    x += dpp_bcast<0x140>(x);
    return x;
}

__launch_bounds__(BLOCK_THREADS)
__global__ void flash_attn_f16_wmma(const float* __restrict__ Q,
                                    const float* __restrict__ K,
                                    const float* __restrict__ V,
                                    float* __restrict__ O) {
    __shared__ __align__(16) _Float16 Klds[KV_TILE][K_STRIDE];       // [key][d]
    __shared__ __align__(16) _Float16 VTlds[DIM][VT_STRIDE];         // [d][key]
    __shared__ __align__(16) _Float16 Plds[WAVES_PER_BLOCK][16][KV_TILE]; // [wave][m][key]

    const int tid  = threadIdx.x;
    const int lane = tid & 31;
    const int wave = tid >> 5;
    const int ln15 = lane & 15;
    const int hi   = (lane & 16) ? 1 : 0;   // upper half-wave?

    const int blocks_per_batch = SEQ / Q_ROWS_PER_BLOCK;   // 64
    const int b  = blockIdx.x / blocks_per_batch;
    const int q0 = (blockIdx.x % blocks_per_batch) * Q_ROWS_PER_BLOCK;

    // scale * log2(e) folded into Q so scores land directly in exp2 domain.
    const float qscale = 0.08838834764831845f * 1.4426950408889634f;

    // ---- Q A-fragments: 16 rows x 128 d, 4 chunks of K=32 ----------------
    const int m_row = q0 + wave * 16 + ln15;
    const float* qrow = Q + ((size_t)b * SEQ + m_row) * DIM;

    H16 qfrag[4];
#pragma unroll
    for (int c = 0; c < 4; ++c) {
        const int dlo = 32 * c + (hi ? 16 : 0);
        const int dhi = dlo + 8;
#pragma unroll
        for (int i = 0; i < 8; ++i) {
            qfrag[c].e[i]     = (_Float16)(qrow[dlo + i] * qscale);
            qfrag[c].e[8 + i] = (_Float16)(qrow[dhi + i] * qscale);
        }
    }

    // O accumulators: C-layout VGPR r -> row m = r + hi*8, col n = 16c+ln15.
    f8 acc[8];
#pragma unroll
    for (int c = 0; c < 8; ++c) acc[c] = (f8)0.0f;

    f8 mstate = (f8)(-1e30f);   // per-row running max (log2 domain)
    f8 lpart  = (f8)0.0f;       // per-lane partial row sums

    // Staging assignments (256 threads).
    const int sk_key  = tid >> 2;          // 0..63 (K rows)
    const int sk_dseg = (tid & 3) * 32;    // 0,32,64,96
    const int sv_kp   = (tid & 31) * 2;    // key pair 0,2,..,62
    const int sv_dseg = (tid >> 5) * 16;   // 0..112

    for (int kv0 = 0; kv0 < SEQ; kv0 += KV_TILE) {
        __syncthreads();  // previous tile fully consumed

        // ---- stage K (row-major f16) ----
        const float* krow = K + ((size_t)b * SEQ + kv0 + sk_key) * DIM + sk_dseg;
#pragma unroll
        for (int i = 0; i < 32; ++i)
            Klds[sk_key][sk_dseg + i] = (_Float16)krow[i];

        // ---- stage V transposed, packed half2 along key ----
        const float* vrow0 = V + ((size_t)b * SEQ + kv0 + sv_kp) * DIM + sv_dseg;
        const float* vrow1 = vrow0 + DIM;
#pragma unroll
        for (int i = 0; i < 16; ++i) {
            h2 pk = { (_Float16)vrow0[i], (_Float16)vrow1[i] };
            *(h2*)&VTlds[sv_dseg + i][sv_kp] = pk;
        }

        // prefetch next tile (global_prefetch_b8) while we compute
        if (kv0 + KV_TILE < SEQ) {
            __builtin_prefetch(krow + (size_t)KV_TILE * DIM, 0, 1);
            __builtin_prefetch(vrow0 + (size_t)KV_TILE * DIM, 0, 1);
        }

        __syncthreads();  // tile visible to all waves

        // ---- S[16x64] = Q[16x128] @ K^T[128x64], 4 key subtiles ----
        f8 s[4];
#pragma unroll
        for (int j = 0; j < 4; ++j) s[j] = (f8)0.0f;
#pragma unroll
        for (int c = 0; c < 4; ++c) {
            const int dbase = 32 * c + (hi ? 16 : 0);
#pragma unroll
            for (int j = 0; j < 4; ++j) {
                H16 bK;
                const _Float16* p = &Klds[16 * j + ln15][dbase];
                bK.h[0] = *(const h8*)p;  bK.h[1] = *(const h8*)(p + 8);
                s[j] = __builtin_amdgcn_wmma_f32_16x16x32_f16(
                    false, qfrag[c].v, false, bK.v, (short)0, s[j], false, false);
            }
        }

        // ---- online softmax (log2 domain; scores are pre-scaled) ----
        f8 corrv;
#pragma unroll
        for (int r = 0; r < 8; ++r) {
            float x0 = s[0][r], x1 = s[1][r], x2 = s[2][r], x3 = s[3][r];
            float mx = fmaxf(fmaxf(x0, x1), fmaxf(x2, x3));
            mx = row16_max(mx);                       // DPP, stays in half-wave
            const float mnew = fmaxf(mstate[r], mx);
            const float p0 = exp2f(x0 - mnew);
            const float p1 = exp2f(x1 - mnew);
            const float p2 = exp2f(x2 - mnew);
            const float p3 = exp2f(x3 - mnew);
            const float corr = exp2f(mstate[r] - mnew);
            mstate[r] = mnew;
            corrv[r]  = corr;
            // per-lane partial sum; cross-lane reduce deferred to the end
            lpart[r] = lpart[r] * corr + ((p0 + p1) + (p2 + p3));

            // spill P (C-layout) into per-wave LDS tile for A-relayout
            const int m = r + (hi ? 8 : 0);
            Plds[wave][m][ln15]      = (_Float16)p0;
            Plds[wave][m][16 + ln15] = (_Float16)p1;
            Plds[wave][m][32 + ln15] = (_Float16)p2;
            Plds[wave][m][48 + ln15] = (_Float16)p3;
        }

        // vector rescale of O accumulators (packs into v_pk_mul_f32)
#pragma unroll
        for (int c = 0; c < 8; ++c) acc[c] *= corrv;

        // per-wave P store -> load ordering + compiler reordering fence
        asm volatile("s_wait_dscnt 0" ::: "memory");

        // reload P in A-fragment layout: two K=32 fragments (keys 0-31, 32-63)
        H16 pA0, pA1;
        {
            const int kb = hi ? 8 : 0;
            pA0.h[0] = *(const h8*)&Plds[wave][ln15][kb];
            pA0.h[1] = *(const h8*)&Plds[wave][ln15][16 + kb];
            pA1.h[0] = *(const h8*)&Plds[wave][ln15][32 + kb];
            pA1.h[1] = *(const h8*)&Plds[wave][ln15][48 + kb];
        }

        // ---- O[16x128] += P[16x64] @ V[64x128], 8 N-chunks x 2 K-halves ----
#pragma unroll
        for (int c = 0; c < 8; ++c) {
            const int drow = 16 * c + ln15;     // n = output d
            const int kb   = hi ? 8 : 0;
            H16 vB0, vB1;
            const _Float16* p = &VTlds[drow][kb];
            vB0.h[0] = *(const h8*)p;        vB0.h[1] = *(const h8*)(p + 16);
            vB1.h[0] = *(const h8*)(p + 32); vB1.h[1] = *(const h8*)(p + 48);
            acc[c] = __builtin_amdgcn_wmma_f32_16x16x32_f16(
                false, pA0.v, false, vB0.v, (short)0, acc[c], false, false);
            acc[c] = __builtin_amdgcn_wmma_f32_16x16x32_f16(
                false, pA1.v, false, vB1.v, (short)0, acc[c], false, false);
        }
    }

    // ---- final cross-lane l reduction, normalize (packed), store O ----
    f8 rinvv;
#pragma unroll
    for (int r = 0; r < 8; ++r) {
        const float l = row16_sum(lpart[r]);
        rinvv[r] = 1.0f / l;
    }
#pragma unroll
    for (int c = 0; c < 8; ++c) acc[c] *= rinvv;

#pragma unroll
    for (int r = 0; r < 8; ++r) {
        const int qrow = q0 + wave * 16 + r + (hi ? 8 : 0);
        float* orow = O + ((size_t)b * SEQ + qrow) * DIM;
#pragma unroll
        for (int c = 0; c < 8; ++c) {
            orow[16 * c + ln15] = acc[c][r];
        }
    }
}

extern "C" void kernel_launch(void* const* d_in, const int* in_sizes, int n_in,
                              void* d_out, int out_size, void* d_ws, size_t ws_size,
                              hipStream_t stream) {
    const float* Q = (const float*)d_in[0];
    const float* K = (const float*)d_in[1];
    const float* V = (const float*)d_in[2];
    float* O = (float*)d_out;
    (void)in_sizes; (void)n_in; (void)out_size; (void)d_ws; (void)ws_size;

    const int grid = BATCH * (SEQ / Q_ROWS_PER_BLOCK);  // 2 * 64 = 128 blocks
    flash_attn_f16_wmma<<<grid, BLOCK_THREADS, 0, stream>>>(Q, K, V, O);
}
